// ImageEncoderViT_1563368096571
// MI455X (gfx1250) — compile-verified, matrix operand order checked by reading, not run on hardware
//
#include <hip/hip_runtime.h>
#include <math.h>

typedef __attribute__((ext_vector_type(16))) __bf16 v16bf;
typedef __attribute__((ext_vector_type(8)))  float  v8f;
typedef __bf16 bf16;
typedef unsigned int u32;
typedef unsigned short u16;

#define TM 64
#define TN 128
#define TK 32

// ---------------------------------------------------------------------------
// Generic batched GEMM: C[M,N] = alpha * A[M,K] * op(B) + bias[N]
//   bT=1: B stored [N,K] row-major (A*B^T, weight layout [out,in])
//   bT=0: B stored [K,N] row-major
// bf16 operands, f32 accumulate via v_wmma_f32_16x16x32_bf16.
// Double-buffered LDS staging, branchless edge handling (clamp+select).
// Batched over gridDim.z = nb1*nb2 with separate strides per batch dim.
// All operand rows are >=8B aligned for every call site (uint2 fast path).
// ---------------------------------------------------------------------------
__global__ __launch_bounds__(256)
void k_gemm(const bf16* __restrict__ A, const bf16* __restrict__ B,
            float* __restrict__ C, bf16* __restrict__ Cb,
            const float* __restrict__ bias,
            int M, int N, int K, int lda, int ldb, int ldc,
            int bT, float alpha, int nb2,
            long sa1, long sa2, long sb1, long sb2, long sc1, long sc2)
{
    const int bz = blockIdx.z;
    const int b1 = bz / nb2, b2 = bz % nb2;
    A += b1 * sa1 + b2 * sa2;
    B += b1 * sb1 + b2 * sb2;
    const long coff = b1 * sc1 + b2 * sc2;
    C += coff;
    if (Cb) Cb += coff;

    __shared__ bf16 As[2][TM][TK + 8];
    __shared__ bf16 Bs[2][TN][TK + 8];

    const int tid  = threadIdx.x;
    const int wave = tid >> 5;          // 0..7
    const int lane = tid & 31;
    const int wm   = wave >> 2;         // 0..1 (M dir)
    const int wn   = wave & 3;          // 0..3 (N dir)
    const int hi16 = lane >> 4;         // lane group (0/1)
    const int l15  = lane & 15;

    const int m0 = blockIdx.y * TM;
    const int n0 = blockIdx.x * TN;

    // staging geometry
    const int ar_ = tid >> 2;           // A row within tile (0..63)
    const int ac_ = (tid & 3) * 8;      // A col within tile (0/8/16/24)
    const int br_ = tid >> 1;           // B row within tile (0..127)
    const int bc_ = (tid & 1) * 16;     // B col within tile (0/16)
    const int gra = m0 + ar_;           // global A row
    const int gnb = n0 + br_;           // global B "n"
    const bool aFullM = (m0 + TM <= M);
    const bool bFullN = (n0 + TN <= N);
    const u16* B16 = (const u16*)B;

    u32 sa[4], sb[8];

    auto stage = [&](int k0) {
        const bool kf = (k0 + TK <= K);
        // ---- A chunk: 8 bf16 per thread ----
        if (aFullM && kf) {
            const uint2* p = (const uint2*)(A + (long)gra * lda + k0 + ac_);
            uint2 q0 = p[0], q1 = p[1];
            sa[0] = q0.x; sa[1] = q0.y; sa[2] = q1.x; sa[3] = q1.y;
        } else {
#pragma unroll
            for (int j = 0; j < 4; ++j) {
                int gk = k0 + ac_ + 2 * j;                   // even, K even
                long idx = (long)(gra < M ? gra : 0) * lda + (gk < K ? gk : 0);
                u32 v = *(const u32*)(A + idx);              // always in-bounds
                if (gra >= M || gk >= K) v = 0;              // v_cndmask
                sa[j] = v;
            }
        }
        // ---- B chunk: 16 bf16 per thread ----
        if (bT) {
            if (bFullN && kf) {
                const uint2* p = (const uint2*)(B + (long)gnb * ldb + k0 + bc_);
                uint2 q0 = p[0], q1 = p[1], q2 = p[2], q3 = p[3];
                sb[0] = q0.x; sb[1] = q0.y; sb[2] = q1.x; sb[3] = q1.y;
                sb[4] = q2.x; sb[5] = q2.y; sb[6] = q3.x; sb[7] = q3.y;
            } else {
#pragma unroll
                for (int j = 0; j < 8; ++j) {
                    int gk = k0 + bc_ + 2 * j;
                    long idx = (long)(gnb < N ? gnb : 0) * ldb + (gk < K ? gk : 0);
                    u32 v = *(const u32*)(B + idx);
                    if (gnb >= N || gk >= K) v = 0;
                    sb[j] = v;
                }
            }
        } else {
            // B stored [K,N]: strided gather, branchless
#pragma unroll
            for (int j = 0; j < 8; ++j) {
                int gk0 = k0 + bc_ + 2 * j, gk1 = gk0 + 1;
                int cn = gnb < N ? gnb : 0;
                u32 v0 = B16[(long)(gk0 < K ? gk0 : 0) * ldb + cn];
                u32 v1 = B16[(long)(gk1 < K ? gk1 : 0) * ldb + cn];
                if (gnb >= N || gk0 >= K) v0 = 0;
                if (gnb >= N || gk1 >= K) v1 = 0;
                sb[j] = v0 | (v1 << 16);
            }
        }
    };

    auto commit = [&](int buf) {
        u32* ap = (u32*)&As[buf][ar_][ac_];
        ap[0] = sa[0]; ap[1] = sa[1]; ap[2] = sa[2]; ap[3] = sa[3];
        u32* bp = (u32*)&Bs[buf][br_][bc_];
#pragma unroll
        for (int j = 0; j < 8; ++j) bp[j] = sb[j];
    };

    v8f acc[2][2] = {};

    stage(0);
    commit(0);
    int buf = 0;
    for (int k0 = 0; k0 < K; k0 += TK) {
        __syncthreads();                       // LDS[buf] ready for all waves
        const bool more = (k0 + TK) < K;
        if (more) stage(k0 + TK);              // global loads overlap wmma below

        // ---- fragments (ISA 7.12.2 16-bit A 16x32 layout) ----
        v16bf afrag[2], bfrag[2];
#pragma unroll
        for (int mi = 0; mi < 2; ++mi) {
            union { int i[8]; v16bf v; } f;
            int ar = wm * 32 + mi * 16 + l15;
#pragma unroll
            for (int vv = 0; vv < 8; ++vv) {
                int kk = 2 * vv + ((vv >= 4) ? 8 : 0) + (hi16 ? 8 : 0);
                f.i[vv] = *(const int*)&As[buf][ar][kk];
            }
            afrag[mi] = f.v;
        }
#pragma unroll
        for (int ni = 0; ni < 2; ++ni) {
            union { int i[8]; v16bf v; } f;
            int br = wn * 32 + ni * 16 + l15;
#pragma unroll
            for (int vv = 0; vv < 8; ++vv) {
                int kk = 2 * vv + ((vv >= 4) ? 8 : 0) + (hi16 ? 8 : 0);
                f.i[vv] = *(const int*)&Bs[buf][br][kk];
            }
            bfrag[ni] = f.v;
        }

#pragma unroll
        for (int mi = 0; mi < 2; ++mi)
#pragma unroll
            for (int ni = 0; ni < 2; ++ni)
                acc[mi][ni] = __builtin_amdgcn_wmma_f32_16x16x32_bf16(
                    false, afrag[mi], false, bfrag[ni],
                    (short)0, acc[mi][ni], false, false);

        if (more) commit(buf ^ 1);             // different buffer: no barrier needed
        buf ^= 1;
    }

    // ---- epilogue: D layout (VGPR r -> M = r + 8*hi16, N = l15) ----
    const int cm  = m0 + wm * 32;
    const int cn0 = n0 + wn * 32;
#pragma unroll
    for (int mi = 0; mi < 2; ++mi) {
#pragma unroll
        for (int ni = 0; ni < 2; ++ni) {
            int col = cn0 + ni * 16 + l15;
            if (col < N) {
                float bv = bias ? bias[col] : 0.0f;
#pragma unroll
                for (int r = 0; r < 8; ++r) {
                    int row = cm + mi * 16 + hi16 * 8 + r;
                    if (row < M) {
                        float v = alpha * acc[mi][ni][r] + bv;
                        C[(long)row * ldc + col] = v;
                        if (Cb) Cb[(long)row * ldc + col] = (bf16)v;
                    }
                }
            }
        }
    }
}

// ---------------------------------------------------------------------------
// Elementwise / data movement kernels
// ---------------------------------------------------------------------------
__global__ void k_convert(bf16* __restrict__ dst, const float* __restrict__ src, long n) {
    long i = (long)blockIdx.x * blockDim.x + threadIdx.x;
    if (i < n) dst[i] = (bf16)src[i];
}

__global__ void k_add(float* __restrict__ out, const float* __restrict__ a,
                      const float* __restrict__ b, long n) {
    long i = (long)blockIdx.x * blockDim.x + threadIdx.x;
    if (i < n) out[i] = a[i] + b[i];
}

__global__ void k_gelu(float* __restrict__ h, long n) {
    long i = (long)blockIdx.x * blockDim.x + threadIdx.x;
    if (i < n) {
        float x = h[i];
        h[i] = 0.5f * x * (1.0f + erff(x * 0.70710678118654752f));
    }
}

// LayerNorm over last dim C, one wave32 per row.
__global__ void k_layernorm(float* __restrict__ out, const float* __restrict__ in,
                            const float* __restrict__ w, const float* __restrict__ b,
                            int rows, int C) {
    int row  = blockIdx.x * 8 + (threadIdx.x >> 5);
    int lane = threadIdx.x & 31;
    if (row >= rows) return;
    const float* p = in + (long)row * C;
    float s = 0.f, s2 = 0.f;
    for (int c = lane; c < C; c += 32) { float v = p[c]; s += v; s2 += v * v; }
    for (int off = 16; off; off >>= 1) {
        s  += __shfl_xor(s,  off, 32);
        s2 += __shfl_xor(s2, off, 32);
    }
    float mean = s / C;
    float var  = s2 / C - mean * mean;
    float rstd = rsqrtf(var + 1e-6f);
    float* q = out + (long)row * C;
    for (int c = lane; c < C; c += 32)
        q[c] = (p[c] - mean) * rstd * w[c] + b[c];
}

// Softmax over length L, one wave32 per row, bf16 output.
__global__ void k_softmax(bf16* __restrict__ out, const float* __restrict__ in,
                          int rows, int L) {
    int row  = blockIdx.x * 8 + (threadIdx.x >> 5);
    int lane = threadIdx.x & 31;
    if (row >= rows) return;
    const float* p = in + (long)row * L;
    float m = -3.0e38f;
    for (int c = lane; c < L; c += 32) m = fmaxf(m, p[c]);
    for (int off = 16; off; off >>= 1) m = fmaxf(m, __shfl_xor(m, off, 32));
    float s = 0.f;
    for (int c = lane; c < L; c += 32) s += __expf(p[c] - m);
    for (int off = 16; off; off >>= 1) s += __shfl_xor(s, off, 32);
    float inv = 1.0f / s;
    bf16* q = out + (long)row * L;
    for (int c = lane; c < L; c += 32) q[c] = (bf16)(__expf(p[c] - m) * inv);
}

// im2col for 16x16/stride16 patch conv: [1024 tokens, 768=c*256+y*16+x] bf16
__global__ void k_im2col_patch(bf16* __restrict__ out, const float* __restrict__ img) {
    long i = (long)blockIdx.x * blockDim.x + threadIdx.x;
    if (i >= (long)1024 * 768) return;
    int k = (int)(i % 768), row = (int)(i / 768);
    int px = row & 31, py = row >> 5;
    int c = k >> 8, r2 = k & 255, yy = r2 >> 4, xx = r2 & 15;
    out[i] = (bf16)img[((long)c * 512 + py * 16 + yy) * 512 + px * 16 + xx];
}

// im2col for 3x3 'SAME' conv over NHWC [32,32,256] -> [1024, 2304=c*9+ky*3+kx]
__global__ void k_im2col3(bf16* __restrict__ out, const float* __restrict__ src) {
    long i = (long)blockIdx.x * blockDim.x + threadIdx.x;
    if (i >= (long)1024 * 2304) return;
    int k = (int)(i % 2304), p = (int)(i / 2304);
    int y = p >> 5, x = p & 31;
    int c = k / 9, r = k % 9, ky = r / 3, kx = r % 3;
    int yy = y + ky - 1, xx = x + kx - 1;
    float v = 0.f;
    if (yy >= 0 && yy < 32 && xx >= 0 && xx < 32) v = src[((long)yy * 32 + xx) * 256 + c];
    out[i] = (bf16)v;
}

// window partition: [1024,768] -> [9*196,768], zero pad (32 -> 42)
__global__ void k_winpart(float* __restrict__ out, const float* __restrict__ in) {
    long i = (long)blockIdx.x * blockDim.x + threadIdx.x;
    if (i >= (long)1764 * 768) return;
    int c = (int)(i % 768), r = (int)(i / 768);
    int win = r / 196, t = r % 196;
    int wy = win / 3, wx = win % 3, iy = t / 14, ix = t % 14;
    int y = wy * 14 + iy, x = wx * 14 + ix;
    out[i] = (y < 32 && x < 32) ? in[((long)y * 32 + x) * 768 + c] : 0.0f;
}

// window unpartition + residual add into X
__global__ void k_unpart_add(float* __restrict__ X, const float* __restrict__ w) {
    long i = (long)blockIdx.x * blockDim.x + threadIdx.x;
    if (i >= (long)1024 * 768) return;
    int c = (int)(i % 768), p = (int)(i / 768);
    int y = p >> 5, x = p & 31;
    int wy = y / 14, iy = y % 14, wx = x / 14, ix = x % 14;
    int row = (wy * 3 + wx) * 196 + iy * 14 + ix;
    X[i] += w[(long)row * 768 + c];
}

// decomposed rel-pos dot products: RH/RW[bh, q, s] = q_vec . rel[(pos - s + S-1)]
__global__ void k_rel_prod(float* __restrict__ RH, float* __restrict__ RW,
                           const bf16* __restrict__ qkvb,
                           const float* __restrict__ relh, const float* __restrict__ relw,
                           int BH, int L, int S, int W, int ldq) {
    long i = (long)blockIdx.x * blockDim.x + threadIdx.x;
    if (i >= (long)BH * L * S) return;
    int s = (int)(i % S);
    int q = (int)((i / S) % L);
    int bh = (int)(i / ((long)S * L));
    int win = bh / 12, head = bh % 12;
    const bf16* qp = qkvb + ((long)win * L + q) * ldq + head * 64;
    int hq = q / W, wq = q % W;
    const float* th = relh + (long)(hq - s + (S - 1)) * 64;
    const float* tw = relw + (long)(wq - s + (S - 1)) * 64;
    float ah = 0.f, aw = 0.f;
#pragma unroll 8
    for (int d = 0; d < 64; ++d) {
        float qv = (float)qp[d];
        ah += qv * th[d];
        aw += qv * tw[d];
    }
    RH[i] = ah;
    RW[i] = aw;
}

// attn[bh,q,k] += RH[bh,q,k/W] + RW[bh,q,k%W]
__global__ void k_rel_add(float* __restrict__ attn, const float* __restrict__ RH,
                          const float* __restrict__ RW, int BH, int L, int S, int W) {
    long i = (long)blockIdx.x * blockDim.x + threadIdx.x;
    if (i >= (long)BH * L * L) return;
    int k = (int)(i % L);
    long rq = i / L;                 // bh*L + q
    attn[i] += RH[rq * S + k / W] + RW[rq * S + k % W];
}

// NHWC [1024,256] -> NCHW [256,1024]
__global__ void k_to_nchw(float* __restrict__ out, const float* __restrict__ in) {
    int i = blockIdx.x * blockDim.x + threadIdx.x;
    if (i >= 256 * 1024) return;
    out[i] = in[(long)(i & 1023) * 256 + (i >> 10)];
}

// ---------------------------------------------------------------------------
// Host orchestration
// ---------------------------------------------------------------------------
extern "C" void kernel_launch(void* const* d_in, const int* in_sizes, int n_in,
                              void* d_out, int out_size, void* d_ws, size_t ws_size,
                              hipStream_t stream) {
    (void)in_sizes; (void)n_in; (void)out_size; (void)ws_size;
    const float* img        = (const float*)d_in[0];
    const float* patch_w    = (const float*)d_in[1];
    const float* patch_b    = (const float*)d_in[2];
    const float* pos_embed  = (const float*)d_in[3];
    const float* norm1_w    = (const float*)d_in[4];
    const float* norm1_b    = (const float*)d_in[5];
    const float* qkv_w      = (const float*)d_in[6];
    const float* qkv_b      = (const float*)d_in[7];
    const float* proj_w     = (const float*)d_in[8];
    const float* proj_b     = (const float*)d_in[9];
    const float* rel_h_win  = (const float*)d_in[10];
    const float* rel_w_win  = (const float*)d_in[11];
    const float* rel_h_glob = (const float*)d_in[12];
    const float* rel_w_glob = (const float*)d_in[13];
    const float* norm2_w    = (const float*)d_in[14];
    const float* norm2_b    = (const float*)d_in[15];
    const float* mlp1_w     = (const float*)d_in[16];
    const float* mlp1_b     = (const float*)d_in[17];
    const float* mlp2_w     = (const float*)d_in[18];
    const float* mlp2_b     = (const float*)d_in[19];
    const float* neck_w1    = (const float*)d_in[20];
    const float* ln1_w      = (const float*)d_in[21];
    const float* ln1_b      = (const float*)d_in[22];
    const float* neck_w2    = (const float*)d_in[23];
    const float* ln2_w      = (const float*)d_in[24];
    const float* ln2_b      = (const float*)d_in[25];

    char* ws = (char*)d_ws;
    size_t off = 0;
    auto alloc = [&](size_t bytes) -> void* {
        void* p = ws + off;
        off = (off + bytes + 255) & ~(size_t)255;
        return p;
    };

    float* X     = (float*)alloc((size_t)1024 * 768 * 4);
    float* LNT   = (float*)alloc((size_t)1024 * 768 * 4);
    float* XW    = (float*)alloc((size_t)1792 * 768 * 4);
    bf16*  XWB   = (bf16*) alloc((size_t)1792 * 768 * 2);
    float* QKV   = (float*)alloc((size_t)1792 * 2304 * 4);
    bf16*  QKVB  = (bf16*) alloc((size_t)1792 * 2304 * 2);
    float* ATT   = (float*)alloc((size_t)12 * 1024 * 1024 * 4);
    bf16*  ATTB  = (bf16*) alloc((size_t)12 * 1024 * 1024 * 2);
    float* AOUT  = (float*)alloc((size_t)1792 * 768 * 4);
    bf16*  AOUTB = (bf16*) alloc((size_t)1792 * 768 * 2);
    float* PROJ  = (float*)alloc((size_t)1792 * 768 * 4);
    float* HID   = (float*)alloc((size_t)1024 * 3072 * 4);
    bf16*  HIDB  = (bf16*) alloc((size_t)1024 * 3072 * 2);
    float* RH    = (float*)alloc((size_t)12 * 1024 * 32 * 4);
    float* RW    = (float*)alloc((size_t)12 * 1024 * 32 * 4);
    bf16*  WB    = (bf16*) alloc((size_t)3072 * 768 * 2);
    bf16*  IM2   = (bf16*) alloc((size_t)1024 * 2304 * 2);
    float* N1    = (float*)alloc((size_t)1024 * 256 * 4);
    float* N2    = (float*)alloc((size_t)1024 * 256 * 4);

    auto nb = [](long n) { return dim3((unsigned)((n + 255) / 256)); };
    auto cvt = [&](bf16* dst, const float* src, long n) {
        k_convert<<<nb(n), 256, 0, stream>>>(dst, src, n);
    };
    auto gemm = [&](const bf16* A, const bf16* B, float* C, bf16* Cb, const float* bias,
                    int M, int N, int K, int lda, int ldb, int ldc, int bT, float alpha,
                    int nb1, int nb2, long sa1, long sa2, long sb1, long sb2,
                    long sc1, long sc2) {
        dim3 g((N + TN - 1) / TN, (M + TM - 1) / TM, nb1 * nb2);
        k_gemm<<<g, 256, 0, stream>>>(A, B, C, Cb, bias, M, N, K, lda, ldb, ldc,
                                      bT, alpha, nb2, sa1, sa2, sb1, sb2, sc1, sc2);
    };

    const long nTok = 1024 * 768;

    // ---- patch embed: im2col + GEMM + pos ----
    k_im2col_patch<<<nb(nTok), 256, 0, stream>>>(IM2, img);
    cvt(WB, patch_w, 768 * 768);
    gemm(IM2, WB, X, nullptr, patch_b, 1024, 768, 768, 768, 768, 768, 1, 1.0f,
         1, 1, 0, 0, 0, 0, 0, 0);
    k_add<<<nb(nTok), 256, 0, stream>>>(X, X, pos_embed, nTok);

    // ---- transformer blocks ----
    for (int i = 0; i < 4; ++i) {
        const bool winb = (i < 3);
        const int L = winb ? 196 : 1024;
        const int S = winb ? 14 : 32;
        const int W = S;
        const int nwin = winb ? 9 : 1;
        const int Mw = winb ? 1764 : 1024;
        const int BH = nwin * 12;

        // LN1
        k_layernorm<<<dim3(128), 256, 0, stream>>>(LNT, X, norm1_w + i * 768,
                                                   norm1_b + i * 768, 1024, 768);
        const float* XWf = LNT;
        if (winb) {
            k_winpart<<<nb((long)1764 * 768), 256, 0, stream>>>(XW, LNT);
            XWf = XW;
        }
        cvt(XWB, XWf, (long)Mw * 768);

        // qkv projection (f32 + bf16 mirror)
        cvt(WB, qkv_w + (long)i * 2304 * 768, (long)2304 * 768);
        gemm(XWB, WB, QKV, QKVB, qkv_b + i * 2304, Mw, 2304, 768,
             768, 768, 2304, 1, 1.0f, 1, 1, 0, 0, 0, 0, 0, 0);

        // attention scores: q * k^T * scale, batched over (win, head)
        gemm(QKVB, QKVB + 768, ATT, nullptr, nullptr, L, L, 64,
             2304, 2304, L, 1, 0.125f,
             nwin, 12, (long)L * 2304, 64, (long)L * 2304, 64,
             (long)12 * L * L, (long)L * L);

        // decomposed rel-pos bias
        const float* relh = winb ? rel_h_win + (long)i * 27 * 64 : rel_h_glob;
        const float* relw = winb ? rel_w_win + (long)i * 27 * 64 : rel_w_glob;
        k_rel_prod<<<nb((long)BH * L * S), 256, 0, stream>>>(
            RH, RW, QKVB, relh, relw, BH, L, S, W, 2304);
        k_rel_add<<<nb((long)BH * L * L), 256, 0, stream>>>(ATT, RH, RW, BH, L, S, W);

        // softmax -> bf16
        int rows = BH * L;
        k_softmax<<<dim3((rows + 7) / 8), 256, 0, stream>>>(ATTB, ATT, rows, L);

        // attn @ v  (B stored [K=L, N=64])
        gemm(ATTB, QKVB + 2 * 768, AOUT, AOUTB, nullptr, L, 64, L,
             L, 2304, 768, 0, 1.0f,
             nwin, 12, (long)12 * L * L, (long)L * L, (long)L * 2304, 64,
             (long)L * 768, 64);

        // output projection
        cvt(WB, proj_w + (long)i * 768 * 768, (long)768 * 768);
        gemm(AOUTB, WB, PROJ, nullptr, proj_b + i * 768, Mw, 768, 768,
             768, 768, 768, 1, 1.0f, 1, 1, 0, 0, 0, 0, 0, 0);

        // residual
        if (winb) k_unpart_add<<<nb(nTok), 256, 0, stream>>>(X, PROJ);
        else      k_add<<<nb(nTok), 256, 0, stream>>>(X, X, PROJ, nTok);

        // MLP
        k_layernorm<<<dim3(128), 256, 0, stream>>>(LNT, X, norm2_w + i * 768,
                                                   norm2_b + i * 768, 1024, 768);
        cvt(XWB, LNT, nTok);
        cvt(WB, mlp1_w + (long)i * 3072 * 768, (long)3072 * 768);
        gemm(XWB, WB, HID, nullptr, mlp1_b + i * 3072, 1024, 3072, 768,
             768, 768, 3072, 1, 1.0f, 1, 1, 0, 0, 0, 0, 0, 0);
        k_gelu<<<nb((long)1024 * 3072), 256, 0, stream>>>(HID, (long)1024 * 3072);
        cvt(HIDB, HID, (long)1024 * 3072);
        cvt(WB, mlp2_w + (long)i * 768 * 3072, (long)768 * 3072);
        gemm(HIDB, WB, PROJ, nullptr, mlp2_b + i * 768, 1024, 768, 3072,
             3072, 3072, 768, 1, 1.0f, 1, 1, 0, 0, 0, 0, 0, 0);
        k_add<<<nb(nTok), 256, 0, stream>>>(X, X, PROJ, nTok);
    }

    // ---- neck ----
    cvt(XWB, X, nTok);
    cvt(WB, neck_w1, 256 * 768);
    gemm(XWB, WB, N1, nullptr, nullptr, 1024, 256, 768, 768, 768, 256, 1, 1.0f,
         1, 1, 0, 0, 0, 0, 0, 0);
    k_layernorm<<<dim3(128), 256, 0, stream>>>(N2, N1, ln1_w, ln1_b, 1024, 256);
    k_im2col3<<<nb((long)1024 * 2304), 256, 0, stream>>>(IM2, N2);
    cvt(WB, neck_w2, (long)256 * 2304);
    gemm(IM2, WB, N1, nullptr, nullptr, 1024, 256, 2304, 2304, 2304, 256, 1, 1.0f,
         1, 1, 0, 0, 0, 0, 0, 0);
    k_layernorm<<<dim3(128), 256, 0, stream>>>(N2, N1, ln2_w, ln2_b, 1024, 256);
    k_to_nchw<<<nb(256 * 1024), 256, 0, stream>>>((float*)d_out, N2);
}
